// Decoder_54958401520086
// MI455X (gfx1250) — matrix operand order repaired and checked by reading, not compiled, not run on hardware
//
#include <hip/hip_runtime.h>

// Problem sizes (fixed by the reference): B=128, L=256, E=256, H=256
#define Bb 128
#define Ll 256
#define Ee 256
#define Hh 256
#define BT 16          // batch rows per workgroup == WMMA M
#define NTH 512        // 16 waves (wave32)

typedef __attribute__((ext_vector_type(16))) __bf16 v16bf;
typedef __attribute__((ext_vector_type(8)))  float  v8f;

union Frag  { v16bf v; uint4 q[2]; };
union Pack8 { uint4 q; __bf16 h[8]; };

__device__ __forceinline__ v8f wmma_bf16(v16bf a, v16bf b, v8f c) {
  // D = A(16x32 bf16) x B(32x16 bf16) + C(16x16 f32)
  return __builtin_amdgcn_wmma_f32_16x16x32_bf16(false, a, false, b,
                                                 (short)0, c, false, false);
}

// A fragment (16x32, 16-bit) per ISA 7.12.2: lane<16 holds M=lane,
// K = kb+8*half + {0..7} in v0..3 and K = kb+16+8*half + {0..7} in v4..7.
__device__ __forceinline__ v16bf load_a_lds(const __bf16* s, int lda,
                                            int kbase, int lane) {
  int hf = lane >> 4;
  int m  = lane & 15;
  Frag f;
  const __bf16* r = s + m * lda + kbase + 8 * hf;
  f.q[0] = *(const uint4*)(r);
  f.q[1] = *(const uint4*)(r + 16);
  return f.v;
}

// B fragment (32x16, 16-bit): lane%16 = N column, element e = K row
// kbase+8*half+{0..7} (e<8) and kbase+16+8*half+{0..7} (e>=8).
// Source is K-major bf16 weight Wt[k][ldn].
__device__ __forceinline__ v16bf load_b_glb(const __bf16* g, int ldn,
                                            int kbase, int nbase, int lane) {
  Frag f;
  int hf = lane >> 4;
  const __bf16* r = g + nbase + (lane & 15);
  #pragma unroll
  for (int e = 0; e < 16; ++e)
    f.v[e] = r[(size_t)(kbase + 8 * hf + e + (e & 8)) * ldn];
  return f.v;
}

__device__ __forceinline__ float sigf(float x) { return 1.f / (1.f + __expf(-x)); }

__device__ __forceinline__ float wred_max(float v) {
  #pragma unroll
  for (int o = 16; o > 0; o >>= 1) v = fmaxf(v, __shfl_xor(v, o, 32));
  return v;
}
__device__ __forceinline__ float wred_sum(float v) {
  #pragma unroll
  for (int o = 16; o > 0; o >>= 1) v += __shfl_xor(v, o, 32);
  return v;
}

// ---------------------------------------------------------------------------
// Kernel 0: convert + transpose weights to bf16, K-major (WMMA B layout)
//   wg_t  [512][1024] : [W_ih ; W_hh]^T   (k<256 -> W_ih, else W_hh)
//   wai_t [256][256]  : W_ai^T
//   wout_t[512][256]  : W_out^T
// ---------------------------------------------------------------------------
__global__ void prep_kernel(const float* __restrict__ W_ih,
                            const float* __restrict__ W_hh,
                            const float* __restrict__ W_ai,
                            const float* __restrict__ W_out,
                            __bf16* __restrict__ wg_t,
                            __bf16* __restrict__ wai_t,
                            __bf16* __restrict__ wout_t) {
  int i = blockIdx.x * blockDim.x + threadIdx.x;
  const int NG = 512 * 1024, NA = 256 * 256, NO = 512 * 256;
  if (i < NG) {
    int k = i >> 10, n = i & 1023;
    float v = (k < Ee) ? W_ih[n * Ee + k] : W_hh[n * Hh + (k - Ee)];
    wg_t[i] = (__bf16)v;
  } else if (i < NG + NA) {
    int j = i - NG; int k = j >> 8, n = j & 255;
    wai_t[j] = (__bf16)W_ai[n * Hh + k];
  } else if (i < NG + NA + NO) {
    int j = i - NG - NA; int k = j >> 8, n = j & 255;
    wout_t[j] = (__bf16)W_out[n * (2 * Hh) + k];
  }
}

// ---------------------------------------------------------------------------
// Kernel 1: ctx[b,l,h] = context[b,l,:] @ W_ac[h,:] + b_ac[h]  (bf16 out)
// One 16x16 output tile per wave; K=256 in 8 steps of 32.
// ---------------------------------------------------------------------------
__global__ void ctx_kernel(const float* __restrict__ context,
                           const float* __restrict__ W_ac,
                           const float* __restrict__ b_ac,
                           __bf16* __restrict__ ctx) {
  int wave = threadIdx.x >> 5, lane = threadIdx.x & 31;
  int tile = blockIdx.x * (blockDim.x >> 5) + wave;  // 0 .. 32767
  int mt = tile >> 4, nt = tile & 15;
  int m0 = mt * 16, n0 = nt * 16;
  int hf = lane >> 4, mr = lane & 15;
  v8f acc;
  #pragma unroll
  for (int r = 0; r < 8; ++r) acc[r] = 0.f;
  for (int kb = 0; kb < Hh; kb += 32) {
    Frag fa, fb;
    const float* arow = context + (size_t)(m0 + mr) * Hh + kb + 8 * hf;
    #pragma unroll
    for (int e = 0; e < 8; ++e) {
      fa.v[e]     = (__bf16)arow[e];        // K = kb + 8*hf + e
      fa.v[8 + e] = (__bf16)arow[16 + e];   // K = kb + 16 + 8*hf + e
    }
    #pragma unroll
    for (int e = 0; e < 16; ++e)            // B[k][n] = W_ac[n][k]
      fb.v[e] = (__bf16)W_ac[(size_t)(n0 + mr) * Hh + kb + 8 * hf + e + (e & 8)];
    acc = wmma_bf16(fa.v, fb.v, acc);
  }
  #pragma unroll
  for (int r = 0; r < 8; ++r) {
    int mm = m0 + r + 8 * hf, nn = n0 + mr;
    ctx[(size_t)mm * Hh + nn] = (__bf16)(acc[r] + b_ac[nn]);
  }
}

// ---------------------------------------------------------------------------
// Kernel 2: persistent fused decoder. One workgroup = 16 batch rows,
// all 256 steps, state resident in LDS. 16 waves, WMMA for all projections.
// ---------------------------------------------------------------------------
__global__ void __launch_bounds__(NTH, 1)
decoder_kernel(const float* __restrict__ embedded,
               const float* __restrict__ dec_in,
               const float* __restrict__ h0,
               const float* __restrict__ c0,
               const float* __restrict__ masksg,
               const float* __restrict__ b_ih,  const float* __restrict__ b_hh,
               const float* __restrict__ b_ai,  const float* __restrict__ b_out,
               const float* __restrict__ Vg,
               const __bf16* __restrict__ ctx,
               const __bf16* __restrict__ wg_t,
               const __bf16* __restrict__ wai_t,
               const __bf16* __restrict__ wout_t,
               float* __restrict__ out_alpha, float* __restrict__ out_ptr,
               float* __restrict__ out_att,
               float* __restrict__ out_h, float* __restrict__ out_c) {
  __shared__ __bf16 sXH[BT][2 * Hh];   // [x | h] / later [hidden | h_t] (A matrix)
  __shared__ float  sC [BT][Hh];       // cell state
  __shared__ float  sHs[BT][Hh];       // h (f32, for final output)
  __shared__ float  sQ [BT][Hh];       // attention query
  __shared__ float  sSc[BT][Ll];       // scores -> alpha (in place)
  __shared__ float  sMask[BT][Ll];     // evolving selection mask
  __shared__ float  sBg[4 * Hh];       // b_ih + b_hh
  __shared__ float  sBai[Hh], sBout[Hh], sV[Hh];
  __shared__ int    sPtr[BT];

  const int tid  = threadIdx.x;
  const int w    = tid >> 5;          // wave 0..15
  const int lane = tid & 31;
  const int hf   = lane >> 4;
  const int mr   = lane & 15;
  const int b0   = blockIdx.x * BT;

  // ---- init state ----
  for (int p = tid; p < BT * Hh; p += NTH) {
    int m = p >> 8, n = p & (Hh - 1);
    float hv = h0[(b0 + m) * Hh + n];
    sHs[m][n] = hv;
    sXH[m][Hh + n] = (__bf16)hv;
    sC[m][n] = c0[(b0 + m) * Hh + n];
    sXH[m][n] = (__bf16)dec_in[(b0 + m) * Ee + n];
  }
  for (int p = tid; p < BT * Ll; p += NTH) {
    int m = p >> 8, l = p & (Ll - 1);
    sMask[m][l] = masksg[(b0 + m) * Ll + l];
  }
  for (int p = tid; p < 4 * Hh; p += NTH) sBg[p] = b_ih[p] + b_hh[p];
  for (int p = tid; p < Hh; p += NTH) {
    sBai[p] = b_ai[p]; sBout[p] = b_out[p]; sV[p] = Vg[p];
  }
  __syncthreads();

  for (int t = 0; t < Ll; ++t) {
    // ---- Phase 1: gates = [x|h] @ Wg^T + bias, fused LSTM cell -----------
    // Wave w owns hidden channels n = 16w+mr: tiles at N = {0,256,512,768}+16w
    // so each lane holds ig/fg/gg/og for its own channels -> fused epilogue.
    v8f g0, g1, g2, g3;
    #pragma unroll
    for (int r = 0; r < 8; ++r) { g0[r] = 0.f; g1[r] = 0.f; g2[r] = 0.f; g3[r] = 0.f; }
    for (int kb = 0; kb < Ee + Hh; kb += 32) {
      v16bf a = load_a_lds(&sXH[0][0], 2 * Hh, kb, lane);
      g0 = wmma_bf16(a, load_b_glb(wg_t, 4 * Hh, kb, 0 * Hh + 16 * w, lane), g0);
      g1 = wmma_bf16(a, load_b_glb(wg_t, 4 * Hh, kb, 1 * Hh + 16 * w, lane), g1);
      g2 = wmma_bf16(a, load_b_glb(wg_t, 4 * Hh, kb, 2 * Hh + 16 * w, lane), g2);
      g3 = wmma_bf16(a, load_b_glb(wg_t, 4 * Hh, kb, 3 * Hh + 16 * w, lane), g3);
    }
    __syncthreads();                       // everyone done reading old h in sXH
    {
      int n = 16 * w + mr;
      #pragma unroll
      for (int r = 0; r < 8; ++r) {
        int m = r + 8 * hf;
        float ig = g0[r] + sBg[0 * Hh + n];
        float fg = g1[r] + sBg[1 * Hh + n];
        float gg = g2[r] + sBg[2 * Hh + n];
        float og = g3[r] + sBg[3 * Hh + n];
        float ct = sigf(fg) * sC[m][n] + sigf(ig) * tanhf(gg);
        float ht = sigf(og) * tanhf(ct);
        sC[m][n] = ct;
        sXH[m][Hh + n] = (__bf16)ht;       // h_t into A-matrix (cols 256..511)
      }
    }
    __syncthreads();

    // ---- Phase 2: q = h_t @ W_ai^T + b_ai --------------------------------
    v8f qa;
    #pragma unroll
    for (int r = 0; r < 8; ++r) qa[r] = 0.f;
    for (int kb = 0; kb < Hh; kb += 32) {
      v16bf a = load_a_lds(&sXH[0][0], 2 * Hh, Hh + kb, lane);
      qa = wmma_bf16(a, load_b_glb(wai_t, Hh, kb, 16 * w, lane), qa);
    }
    {
      int n = 16 * w + mr;
      #pragma unroll
      for (int r = 0; r < 8; ++r) sQ[r + 8 * hf][n] = qa[r] + sBai[n];
    }
    __syncthreads();

    // ---- Phase 3: scores[m,l] = sum_h V[h]*tanh(q[m,h] + ctx[b,l,h]) -----
    for (int p = tid; p < BT * Ll; p += NTH) {
      int m = p >> 8, l = p & (Ll - 1);
      const __bf16* cr = ctx + ((size_t)((b0 + m) * Ll + l)) * Hh;
      float s = 0.f;
      for (int hb = 0; hb < Hh; hb += 8) {
        Pack8 c8; c8.q = *(const uint4*)(cr + hb);
        #pragma unroll
        for (int j = 0; j < 8; ++j)
          s += sV[hb + j] * tanhf(sQ[m][hb + j] + (float)c8.h[j]);
      }
      sSc[m][l] = s;
    }
    __syncthreads();

    // ---- Phase 4: softmaxes, pointer, mask update (one wave per row) -----
    {
      int m = w;
      int bg = b0 + m;
      float sv = masksg[bg * Ll + t];                  // step_valid
      float vals[8], vm[8], vd[8], ex[8];
      #pragma unroll
      for (int j = 0; j < 8; ++j) {
        int l = lane + 32 * j;
        vals[j] = sSc[m][l];
        vm[j]   = sMask[m][l];
        vd[j]   = masksg[bg * Ll + l];                 // length-valid
      }
      // attn = softmax over valid
      float mx = -1e30f;
      #pragma unroll
      for (int j = 0; j < 8; ++j) mx = fmaxf(mx, vd[j] > 0.f ? vals[j] : -1e30f);
      mx = wred_max(mx);
      float sum = 0.f;
      #pragma unroll
      for (int j = 0; j < 8; ++j) { ex[j] = vd[j] > 0.f ? __expf(vals[j] - mx) : 0.f; sum += ex[j]; }
      sum = wred_sum(sum);
      float inv = 1.f / sum;
      #pragma unroll
      for (int j = 0; j < 8; ++j) {
        int l = lane + 32 * j;
        out_att[((size_t)bg * Ll + t) * Ll + l] = ex[j] * inv * sv;
      }
      // alpha = softmax over sel (fallback to valid if row exhausted)
      float any = 0.f;
      #pragma unroll
      for (int j = 0; j < 8; ++j) any += vm[j];
      any = wred_sum(any);
      bool has = any > 0.f;
      float mx2 = -1e30f;
      float keep[8];
      #pragma unroll
      for (int j = 0; j < 8; ++j) {
        bool sel = has ? (vm[j] > 0.f) : (vd[j] > 0.f);
        keep[j] = sel ? 1.f : 0.f;
        mx2 = fmaxf(mx2, sel ? vals[j] : -1e30f);
      }
      mx2 = wred_max(mx2);
      float s2 = 0.f;
      #pragma unroll
      for (int j = 0; j < 8; ++j) { ex[j] = keep[j] > 0.f ? __expf(vals[j] - mx2) : 0.f; s2 += ex[j]; }
      s2 = wred_sum(s2);
      float inv2 = 1.f / s2;
      float bestv = -1.f; int bestl = 0;
      #pragma unroll
      for (int j = 0; j < 8; ++j) {
        int l = lane + 32 * j;
        float al = ex[j] * inv2;
        sSc[m][l] = al;                                  // alpha in place
        out_alpha[((size_t)bg * Ll + t) * Ll + l] = al * sv;
        float pv = al * vm[j];
        if (pv > bestv) { bestv = pv; bestl = l; }
      }
      #pragma unroll
      for (int o = 16; o > 0; o >>= 1) {
        float ov = __shfl_xor(bestv, o, 32);
        int   ol = __shfl_xor(bestl, o, 32);
        if (ov > bestv || (ov == bestv && ol < bestl)) { bestv = ov; bestl = ol; }
      }
      if (lane == 0) {
        sPtr[m] = bestl;
        out_ptr[bg * Ll + t] = sv > 0.f ? (float)bestl : 0.f;
        sMask[m][bestl] = 0.f;
      }
    }
    __syncthreads();

    // ---- Phase 5: hidden[m,h] = sum_l alpha[m,l]*ctx[b,l,h] --------------
    for (int p = tid; p < BT * Hh; p += NTH) {
      int m = p >> 8, n = p & (Hh - 1);
      const __bf16* cb = ctx + (size_t)(b0 + m) * Ll * Hh + n;
      float acc = 0.f;
      for (int l = 0; l < Ll; ++l)
        acc += sSc[m][l] * (float)cb[(size_t)l * Hh];
      sXH[m][n] = (__bf16)acc;             // hidden into A-matrix cols 0..255
    }
    __syncthreads();

    // ---- Phase 6: h_new = tanh([hidden|h_t] @ W_out^T + b_out) -----------
    v8f oa;
    #pragma unroll
    for (int r = 0; r < 8; ++r) oa[r] = 0.f;
    for (int kb = 0; kb < 2 * Hh; kb += 32) {
      v16bf a = load_a_lds(&sXH[0][0], 2 * Hh, kb, lane);
      oa = wmma_bf16(a, load_b_glb(wout_t, Hh, kb, 16 * w, lane), oa);
    }
    __syncthreads();                       // all reads of sXH done before rewrite
    {
      int n = 16 * w + mr;
      #pragma unroll
      for (int r = 0; r < 8; ++r) {
        int m = r + 8 * hf;
        float hn = tanhf(oa[r] + sBout[n]);
        sHs[m][n] = hn;
        sXH[m][Hh + n] = (__bf16)hn;       // next-step h
      }
    }
    // ---- Phase 7: x_new = embedded[b, ptr, :] ----------------------------
    for (int p = tid; p < BT * Ee; p += NTH) {
      int m = p >> 8, e = p & (Ee - 1);
      sXH[m][e] = (__bf16)embedded[((size_t)((b0 + m) * Ll + sPtr[m])) * Ee + e];
    }
    __syncthreads();
  }

  // ---- final hT, cT ----
  for (int p = tid; p < BT * Hh; p += NTH) {
    int m = p >> 8, n = p & (Hh - 1);
    out_h[(b0 + m) * Hh + n] = sHs[m][n];
    out_c[(b0 + m) * Hh + n] = sC[m][n];
  }
}

// ---------------------------------------------------------------------------
extern "C" void kernel_launch(void* const* d_in, const int* in_sizes, int n_in,
                              void* d_out, int out_size, void* d_ws, size_t ws_size,
                              hipStream_t stream) {
  const float* embedded = (const float*)d_in[0];
  const float* dec_in   = (const float*)d_in[1];
  const float* h0       = (const float*)d_in[2];
  const float* c0       = (const float*)d_in[3];
  const float* context  = (const float*)d_in[4];
  const float* masks    = (const float*)d_in[5];
  const float* W_ih     = (const float*)d_in[6];
  const float* b_ih     = (const float*)d_in[7];
  const float* W_hh     = (const float*)d_in[8];
  const float* b_hh     = (const float*)d_in[9];
  const float* W_out    = (const float*)d_in[10];
  const float* b_out    = (const float*)d_in[11];
  const float* W_ai     = (const float*)d_in[12];
  const float* b_ai     = (const float*)d_in[13];
  const float* W_ac     = (const float*)d_in[14];
  const float* b_ac     = (const float*)d_in[15];
  const float* Vv       = (const float*)d_in[16];

  // workspace carve-out (bf16 ctx + K-major bf16 weights)
  char* ws = (char*)d_ws;
  __bf16* ctx    = (__bf16*)ws;                 ws += (size_t)Bb * Ll * Hh * 2;
  __bf16* wg_t   = (__bf16*)ws;                 ws += (size_t)512 * 1024 * 2;
  __bf16* wai_t  = (__bf16*)ws;                 ws += (size_t)256 * 256 * 2;
  __bf16* wout_t = (__bf16*)ws;                 ws += (size_t)512 * 256 * 2;

  prep_kernel<<<(512 * 1024 + 256 * 256 + 512 * 256) / 256, 256, 0, stream>>>(
      W_ih, W_hh, W_ai, W_out, wg_t, wai_t, wout_t);

  // (B*L/16) * (H/16) = 32768 wave-tiles, 4 waves per block
  ctx_kernel<<<32768 / 4, 128, 0, stream>>>(context, W_ac, b_ac, ctx);

  float* out = (float*)d_out;
  const size_t OPTR = (size_t)Bb * Ll * Ll;          // outputs
  const size_t OATT = OPTR + (size_t)Bb * Ll;        // pointers
  const size_t OH   = OATT + (size_t)Bb * Ll * Ll;   // atts
  const size_t OC   = OH + (size_t)Bb * Hh;          // hT

  decoder_kernel<<<Bb / BT, NTH, 0, stream>>>(
      embedded, dec_in, h0, c0, masks, b_ih, b_hh, b_ai, b_out, Vv,
      ctx, wg_t, wai_t, wout_t,
      out, out + OPTR, out + OATT, out + OH, out + OC);
}
